// CausalSelfAttention_46282567581814
// MI455X (gfx1250) — matrix operand order repaired and checked by reading, not compiled, not run on hardware
//
#include <hip/hip_runtime.h>

typedef __attribute__((ext_vector_type(16))) __bf16 v16bf;
typedef __attribute__((ext_vector_type(8)))  __bf16 v8bf;
typedef __attribute__((ext_vector_type(4)))  __bf16 v4bf;
typedef __attribute__((ext_vector_type(8)))  float  v8f;
typedef __attribute__((ext_vector_type(4)))  int    v4i;

#define BATCH 2
#define TSEQ  2048
#define CDIM  1024
#define NHEAD 16
#define HDIM  64
#define MROWS (BATCH*TSEQ)   /* 4096 */
#define N_QKV (3*CDIM)       /* 3072 */

// ---------------- CDNA5 feature detection (toolchain-portable) ----------------
#define TDM_HDR_PRESENT 0
#if defined(__has_include)
#  if __has_include(<hip/amd_detail/amd_gfx1250_TDM.h>)
#    undef TDM_HDR_PRESENT
#    define TDM_HDR_PRESENT 1
#  endif
#endif

#if defined(__has_builtin)
#  if __has_builtin(__builtin_amdgcn_global_load_async_to_lds_b128)
#    define HAVE_ASYNC_LDS 1
#  endif
#  if __has_builtin(__builtin_amdgcn_tensor_load_to_lds) && !TDM_HDR_PRESENT
#    define HAVE_TDM5 1   /* ROCm 7.2 5-arg form */
#  endif
#endif
#ifndef HAVE_ASYNC_LDS
#  define HAVE_ASYNC_LDS 0
#endif
#ifndef HAVE_TDM5
#  define HAVE_TDM5 0
#endif

#define AS1 __attribute__((address_space(1)))
#define AS3 __attribute__((address_space(3)))

// 16-byte global -> LDS copy; async DMA when available, VGPR bounce otherwise.
// Builtin signature (from toolchain diagnostic): (v4i AS1*, v4i AS3*, Ii, Ii).
static __device__ __forceinline__ void async_cp16(__bf16* lds, const __bf16* g) {
#if HAVE_ASYNC_LDS
  __builtin_amdgcn_global_load_async_to_lds_b128(
      (AS1 v4i*)(unsigned long long)(__SIZE_TYPE__)g,
      (AS3 v4i*)(unsigned)(__SIZE_TYPE__)lds, 0, 0);
#else
  *(v8bf*)lds = *(const v8bf*)g;
#endif
}

static __device__ __forceinline__ void async_wait() {
#if HAVE_ASYNC_LDS
#  if defined(__has_builtin) && __has_builtin(__builtin_amdgcn_s_wait_asynccnt)
  __builtin_amdgcn_s_wait_asynccnt(0);
#  else
  asm volatile("s_wait_asynccnt 0x0" ::: "memory");
#  endif
#endif
}

#if HAVE_TDM5
typedef unsigned int u32x4 __attribute__((ext_vector_type(4)));
typedef int          i32x8 __attribute__((ext_vector_type(8)));
typedef int          i32x4 __attribute__((ext_vector_type(4)));

// 1-D contiguous TDM copy of nelem bf16 elements, global -> LDS.
// D# per CDNA5 ISA: group0 {count=1, lds_addr,
// global_addr[56:0], type=2}; group1 {data_size=2B, tensor_dim0=nelem,
// tensor_dim1=1, tile_dim0=nelem, tile_dim1=1, dim0_stride=nelem}.
static __device__ __forceinline__ void tdm_load_1d(__bf16* lds, const __bf16* g, int nelem) {
  unsigned long long ga = (unsigned long long)(__SIZE_TYPE__)g;
  unsigned la = (unsigned)(__SIZE_TYPE__)lds;   // flat LDS addr[31:0] == LDS offset
  u32x4 g0;
  g0[0] = 1u;                                                   // count=1
  g0[1] = la;                                                   // lds_addr
  g0[2] = (unsigned)(ga & 0xffffffffull);                       // global_addr[31:0]
  g0[3] = (unsigned)((ga >> 32) & 0x1ffffffull) | (2u << 30);   // [56:32] | type=2
  i32x8 g1;
  g1[0] = (int)(1u << 16);                                      // data_size=2B
  g1[1] = (int)(((unsigned)nelem & 0xffffu) << 16);             // tensor_dim0 lo16
  g1[2] = (int)((((unsigned)nelem >> 16) & 0xffffu) | (1u << 16)); // dim0 hi16, dim1=1
  g1[3] = (int)(((unsigned)nelem & 0xffffu) << 16);             // tile_dim0
  g1[4] = 1;                                                    // tile_dim1 = 1
  g1[5] = nelem;                                                // dim0_stride lo32
  g1[6] = 0;
  g1[7] = 0;
  i32x4 zz; zz[0] = 0; zz[1] = 0; zz[2] = 0; zz[3] = 0;
  __builtin_amdgcn_tensor_load_to_lds(g0, g1, zz, zz, 0);
}
#endif

// ---------------- WMMA helpers (layouts per CDNA5 ISA 7.12.2) -----
static __device__ __forceinline__ v8f wmma_bf16(v16bf a, v16bf b, v8f c) {
  return __builtin_amdgcn_wmma_f32_16x16x32_bf16(false, a, false, b, (short)0, c, false, false);
}

static __device__ __forceinline__ v8f zero8() {
  v8f z;
#pragma unroll
  for (int i = 0; i < 8; ++i) z[i] = 0.0f;
  return z;
}

static __device__ __forceinline__ v16bf combine16(v8bf lo, v8bf hi) {
  v16bf r;
#pragma unroll
  for (int i = 0; i < 8; ++i) { r[i] = lo[i]; r[i + 8] = hi[i]; }
  return r;
}

// A-fragment (16x32 bf16, MxK) from K-contiguous LDS tile [rows][ldk].
static __device__ __forceinline__ v16bf load_a_frag(const __bf16* base, int ldk, int row0, int kb) {
  int lane = threadIdx.x & 31;
  int m  = row0 + (lane & 15);
  int ks = kb + ((lane & 16) ? 8 : 0);
  const __bf16* p = base + m * ldk + ks;
  return combine16(*(const v8bf*)p, *(const v8bf*)(p + 16));
}

// B-fragment (32x16 bf16, KxN) from LDS tile stored [n][k] (k contiguous).
static __device__ __forceinline__ v16bf load_b_frag(const __bf16* base, int ldk, int n0, int kb) {
  int lane = threadIdx.x & 31;
  int n  = n0 + (lane & 15);
  int ks = kb + ((lane & 16) ? 8 : 0);
  const __bf16* p = base + n * ldk + ks;
  return combine16(*(const v8bf*)p, *(const v8bf*)(p + 16));
}

__global__ __launch_bounds__(256) void cast_f32_bf16(const float* __restrict__ in,
                                                     __bf16* __restrict__ out, int n4) {
  int i = blockIdx.x * blockDim.x + threadIdx.x;
  if (i < n4) {
    float4 f = ((const float4*)in)[i];
    v4bf o;
    o[0] = (__bf16)f.x; o[1] = (__bf16)f.y; o[2] = (__bf16)f.z; o[3] = (__bf16)f.w;
    ((v4bf*)out)[i] = o;
  }
}

// Tiled GEMM: C = A[M,K] * B[K,N] + bias, bf16 in, fp32 accumulate.
// 128x128 block tile, BK=64 (two WMMA K-steps per barrier pair).
// MODE 1: fp32 row-major to outF. MODE 0: scatter into bf16 Q/K/V [B,NH,T,HD].
template <int MODE>
__global__ __launch_bounds__(256) void gemm_bf16_wmma(
    const __bf16* __restrict__ A,
    const __bf16* __restrict__ Bm,
    const float*  __restrict__ bias,
    float*        __restrict__ outF,
    __bf16* __restrict__ Qp, __bf16* __restrict__ Kp, __bf16* __restrict__ Vp,
    int M, int N, int K)
{
  __shared__ alignas(16) __bf16 Alds[128 * 64];  // [m][k]
  __shared__ alignas(16) __bf16 Blds[128 * 64];  // [n][k]

  const int tid  = threadIdx.x;
  const int lane = tid & 31;
  const int lo   = lane & 15;
  const int hi   = (lane >> 4) & 1;
  const int wave = tid >> 5;
  const int wm = (wave >> 2) * 64;   // 2 waves along M
  const int wn = (wave & 3) * 32;    // 4 waves along N
  const int rowBase = blockIdx.y * 128;
  const int nBase   = blockIdx.x * 128;

  v8f acc[4][2];
#pragma unroll
  for (int i = 0; i < 4; ++i)
#pragma unroll
    for (int j = 0; j < 2; ++j) acc[i][j] = zero8();

  for (int k0 = 0; k0 < K; k0 += 64) {
    // A tile 128x64 halfs: layout-preserving -> async DMA to LDS
#pragma unroll
    for (int r = 0; r < 4; ++r) {
      int idx = r * 256 + tid;
      int row = idx >> 3;
      int kc  = (idx & 7) * 8;
      async_cp16(&Alds[row * 64 + kc],
                 &A[(size_t)(rowBase + row) * K + k0 + kc]);
    }
    // B tile 64x128 -> element transpose into [n][k] (VGPR scatter)
#pragma unroll
    for (int r = 0; r < 4; ++r) {
      int idx = r * 256 + tid;
      int kk  = idx >> 4;          // 0..63
      int n8  = (idx & 15) * 8;
      v8bf t = *(const v8bf*)&Bm[(size_t)(k0 + kk) * N + nBase + n8];
#pragma unroll
      for (int i = 0; i < 8; ++i) Blds[(n8 + i) * 64 + kk] = t[i];
    }
    if (k0 + 64 < K)
      __builtin_prefetch(&Bm[(size_t)(k0 + 64 + (tid >> 2)) * N + nBase + (tid & 3) * 32], 0, 0);
    async_wait();
    __syncthreads();

#pragma unroll
    for (int kb = 0; kb < 64; kb += 32) {
      v16bf af[4], bfr[2];
#pragma unroll
      for (int fm = 0; fm < 4; ++fm) af[fm] = load_a_frag(Alds, 64, wm + fm * 16, kb);
#pragma unroll
      for (int fn = 0; fn < 2; ++fn) bfr[fn] = load_b_frag(Blds, 64, wn + fn * 16, kb);
#pragma unroll
      for (int fm = 0; fm < 4; ++fm)
#pragma unroll
        for (int fn = 0; fn < 2; ++fn)
          acc[fm][fn] = wmma_bf16(af[fm], bfr[fn], acc[fm][fn]);
    }
    __syncthreads();
  }

  // Epilogue. C/D layout: VGPR r -> row = r + 8*hi, col = lane&15.
#pragma unroll
  for (int fm = 0; fm < 4; ++fm)
#pragma unroll
    for (int fn = 0; fn < 2; ++fn)
#pragma unroll
      for (int r = 0; r < 8; ++r) {
        int row = rowBase + wm + fm * 16 + r + hi * 8;
        int col = nBase + wn + fn * 16 + lo;
        float val = acc[fm][fn][r] + bias[col];
        if (MODE == 1) {
          outF[(size_t)row * N + col] = val;
        } else {
          int which = col >> 10;            // 0:q 1:k 2:v
          int c  = col & (CDIM - 1);
          int hh = c >> 6;
          int dd = c & (HDIM - 1);
          int bb = row >> 11;               // / TSEQ
          int tt = row & (TSEQ - 1);
          size_t dst = (((size_t)bb * NHEAD + hh) * TSEQ + tt) * HDIM + dd;
          __bf16* p = (which == 0) ? Qp : (which == 1) ? Kp : Vp;
          p[dst] = (__bf16)val;
        }
      }
}

// Flash attention: one workgroup per (b, h, 64-query tile); 4 waves x 16 rows.
__global__ __launch_bounds__(128) void flash_attn_wmma(
    const __bf16* __restrict__ Qp,
    const __bf16* __restrict__ Kp,
    const __bf16* __restrict__ Vp,
    __bf16* __restrict__ Yp)
{
  __shared__ alignas(16) __bf16 Qlds[64 * 64];     // [m][d]
  __shared__ alignas(16) __bf16 Klds[64 * 64];     // [key][d]  ([n][k] for B-frag)
  __shared__ alignas(16) __bf16 Vlds[64 * 64];     // [d][key]  ([n][k] for B-frag)
  __shared__ alignas(16) __bf16 Plds[4][16 * 64];  // per-wave P [m][key]

  const int ntiles = TSEQ / 64;                    // 32
  const int bid = blockIdx.x;
  const int qt  = bid % ntiles;
  const int h   = (bid / ntiles) % NHEAD;
  const int b   = bid / (ntiles * NHEAD);
  const int q0  = qt * 64;

  const int tid  = threadIdx.x;
  const int wave = tid >> 5;
  const int lane = tid & 31;
  const int lo   = lane & 15;
  const int hi   = (lane >> 4) & 1;

  const size_t headOff = ((size_t)(b * NHEAD + h)) * TSEQ * HDIM;
  const __bf16* qbase = Qp + headOff + (size_t)q0 * HDIM;

  // Q tile (8 KB contiguous): TDM when available, else async copies.
#if HAVE_TDM5
  if (wave == 0) {
    tdm_load_1d(Qlds, qbase, 64 * HDIM);
    __builtin_amdgcn_s_wait_tensorcnt(0);
  }
#else
#pragma unroll
  for (int r = 0; r < 4; ++r) {
    int idx = r * 128 + tid;
    int m  = idx >> 3;
    int d8 = (idx & 7) * 8;
    async_cp16(&Qlds[m * 64 + d8], &qbase[m * 64 + d8]);
  }
  async_wait();
#endif

  v8f o[4];
#pragma unroll
  for (int fd = 0; fd < 4; ++fd) o[fd] = zero8();
  float mrow[8], lrow[8];
#pragma unroll
  for (int r = 0; r < 8; ++r) { mrow[r] = -__builtin_inff(); lrow[r] = 0.0f; }

  for (int j = 0; j <= qt; ++j) {
    __syncthreads();  // all waves done reading previous K/V tiles (and Q ready)
    const __bf16* kb = Kp + headOff + (size_t)j * 64 * HDIM;
    const __bf16* vb = Vp + headOff + (size_t)j * 64 * HDIM;
    // K tile: layout-preserving -> async DMA; V tile: transpose via VGPRs.
#pragma unroll
    for (int r = 0; r < 4; ++r) {
      int idx = r * 128 + tid;
      int key = idx >> 3;
      int d8  = (idx & 7) * 8;
      async_cp16(&Klds[key * 64 + d8], &kb[key * 64 + d8]);
    }
#pragma unroll
    for (int r = 0; r < 4; ++r) {
      int idx = r * 128 + tid;
      int key = idx >> 3;
      int d8  = (idx & 7) * 8;
      v8bf vv = *(const v8bf*)&vb[key * 64 + d8];
#pragma unroll
      for (int i = 0; i < 8; ++i) Vlds[(d8 + i) * 64 + key] = vv[i];
    }
    async_wait();
    __syncthreads();

    // S = Q * K^T  (16x64 per wave, contraction over d=64, two WMMA steps)
    v16bf aq0 = load_a_frag(Qlds, 64, wave * 16, 0);
    v16bf aq1 = load_a_frag(Qlds, 64, wave * 16, 32);
    v8f s[4];
#pragma unroll
    for (int f = 0; f < 4; ++f) {
      v8f z = zero8();
      z = wmma_bf16(aq0, load_b_frag(Klds, 64, f * 16, 0), z);
      z = wmma_bf16(aq1, load_b_frag(Klds, 64, f * 16, 32), z);
      s[f] = z;
    }
    const float sc = 0.125f;  // HD^-0.5
#pragma unroll
    for (int f = 0; f < 4; ++f)
#pragma unroll
      for (int r = 0; r < 8; ++r) s[f][r] *= sc;
    if (j == qt) {  // causal mask only on the diagonal tile
#pragma unroll
      for (int f = 0; f < 4; ++f)
#pragma unroll
        for (int r = 0; r < 8; ++r) {
          int qg = wave * 16 + r + hi * 8;
          int kg = f * 16 + lo;
          if (kg > qg) s[f][r] = -__builtin_inff();
        }
    }

    // Online softmax. Row = (r, hi-half); reduce over f and the 16-lane half.
    float rmax[8], scl[8], psum[8];
#pragma unroll
    for (int r = 0; r < 8; ++r) {
      float v = s[0][r];
#pragma unroll
      for (int f = 1; f < 4; ++f) v = fmaxf(v, s[f][r]);
#pragma unroll
      for (int msk = 1; msk < 16; msk <<= 1) v = fmaxf(v, __shfl_xor(v, msk, 32));
      rmax[r] = v;
    }
#pragma unroll
    for (int r = 0; r < 8; ++r) {
      float mn = fmaxf(mrow[r], rmax[r]);
      scl[r] = __expf(mrow[r] - mn);
      mrow[r] = mn;
      psum[r] = 0.0f;
    }
#pragma unroll
    for (int f = 0; f < 4; ++f)
#pragma unroll
      for (int r = 0; r < 8; ++r) {
        float p = __expf(s[f][r] - mrow[r]);  // masked -> exp(-inf)=0
        psum[r] += p;
        Plds[wave][(r + hi * 8) * 64 + f * 16 + lo] = (__bf16)p;
      }
#pragma unroll
    for (int r = 0; r < 8; ++r) {
      float v = psum[r];
#pragma unroll
      for (int msk = 1; msk < 16; msk <<= 1) v += __shfl_xor(v, msk, 32);
      lrow[r] = lrow[r] * scl[r] + v;
    }
#pragma unroll
    for (int fd = 0; fd < 4; ++fd)
#pragma unroll
      for (int r = 0; r < 8; ++r) o[fd][r] *= scl[r];

    // Wave-local P stores must land before re-reading as A-fragments.
    asm volatile("s_wait_dscnt 0x0" ::: "memory");

    // O += P * V  (contraction over 64 keys, two WMMA steps)
    v16bf ap0 = load_a_frag(&Plds[wave][0], 64, 0, 0);
    v16bf ap1 = load_a_frag(&Plds[wave][0], 64, 0, 32);
#pragma unroll
    for (int fd = 0; fd < 4; ++fd) {
      o[fd] = wmma_bf16(ap0, load_b_frag(Vlds, 64, fd * 16, 0), o[fd]);
      o[fd] = wmma_bf16(ap1, load_b_frag(Vlds, 64, fd * 16, 32), o[fd]);
    }
  }

  // Write y tile back as [B, T, C] bf16 (fused head transpose).
#pragma unroll
  for (int fd = 0; fd < 4; ++fd)
#pragma unroll
    for (int r = 0; r < 8; ++r) {
      int m = wave * 16 + r + hi * 8;
      int t = q0 + m;
      float val = o[fd][r] / lrow[r];
      Yp[((size_t)(b * TSEQ + t)) * CDIM + h * HDIM + fd * 16 + lo] = (__bf16)val;
    }
}

extern "C" void kernel_launch(void* const* d_in, const int* in_sizes, int n_in,
                              void* d_out, int out_size, void* d_ws, size_t ws_size,
                              hipStream_t stream)
{
  const float* x     = (const float*)d_in[0];
  const float* Wqkv  = (const float*)d_in[1];
  const float* bqkv  = (const float*)d_in[2];
  const float* Wproj = (const float*)d_in[3];
  const float* bproj = (const float*)d_in[4];
  float* out = (float*)d_out;

  char* ws = (char*)d_ws;
  size_t off = 0;
  __bf16* xb     = (__bf16*)(ws + off); off += (size_t)MROWS * CDIM * 2;   // 8 MB
  __bf16* wqkvb  = (__bf16*)(ws + off); off += (size_t)CDIM * N_QKV * 2;   // 6 MB
  __bf16* wprojb = (__bf16*)(ws + off); off += (size_t)CDIM * CDIM * 2;    // 2 MB
  __bf16* Qb     = (__bf16*)(ws + off); off += (size_t)MROWS * CDIM * 2;   // 8 MB
  __bf16* Kb     = (__bf16*)(ws + off); off += (size_t)MROWS * CDIM * 2;   // 8 MB
  __bf16* Vb     = (__bf16*)(ws + off); off += (size_t)MROWS * CDIM * 2;   // 8 MB
  __bf16* Yb     = (__bf16*)(ws + off); off += (size_t)MROWS * CDIM * 2;   // 8 MB

  // 1) fp32 -> bf16 casts
  {
    int n4 = MROWS * CDIM / 4;
    cast_f32_bf16<<<(n4 + 255) / 256, 256, 0, stream>>>(x, xb, n4);
    n4 = CDIM * N_QKV / 4;
    cast_f32_bf16<<<(n4 + 255) / 256, 256, 0, stream>>>(Wqkv, wqkvb, n4);
    n4 = CDIM * CDIM / 4;
    cast_f32_bf16<<<(n4 + 255) / 256, 256, 0, stream>>>(Wproj, wprojb, n4);
  }

  // 2) QKV projection, scattered into head-major Q/K/V
  gemm_bf16_wmma<0><<<dim3(N_QKV / 128, MROWS / 128), 256, 0, stream>>>(
      xb, wqkvb, bqkv, nullptr, Qb, Kb, Vb, MROWS, N_QKV, CDIM);

  // 3) causal flash attention
  flash_attn_wmma<<<BATCH * NHEAD * (TSEQ / 64), 128, 0, stream>>>(Qb, Kb, Vb, Yb);

  // 4) output projection -> fp32 d_out
  gemm_bf16_wmma<1><<<dim3(CDIM / 128, MROWS / 128), 256, 0, stream>>>(
      Yb, wprojb, bproj, out, nullptr, nullptr, nullptr, MROWS, CDIM, CDIM);
}
